// NNUE_27281632264596
// MI455X (gfx1250) — compile-verified, matrix operand order checked by reading, not run on hardware
//
#include <hip/hip_runtime.h>

// NNUE forward for MI455X (gfx1250, wave32).
// Roofline: mandatory read of dense multihot inputs = 1.34 GB ~= 58 us @ 23.3 TB/s.
// Move head = 9.7 GFLOP of bf16 WMMA + 67 MB write; B-tiles staged in LDS via
// GLOBAL_LOAD_ASYNC_TO_LDS_B128 (ASYNCcnt-tracked) so L2 traffic stays ~64 MB.

#define HALF_KP 40960
#define BATCH   4096
#define MOVE_N  4096

typedef __attribute__((ext_vector_type(16))) __bf16 v16bf;
typedef __attribute__((ext_vector_type(8)))  float  v8f;
typedef __attribute__((ext_vector_type(4)))  int    v4i_t;
typedef __attribute__((address_space(1))) v4i_t* g_v4i_p;   // global
typedef __attribute__((address_space(3))) v4i_t* l_v4i_p;   // LDS

// float -> bf16, round-to-nearest-even (bit-exact, no header dependence)
static __device__ __forceinline__ unsigned short f2bf_u(float f) {
  unsigned int u = __float_as_uint(f);
  u += 0x7fffu + ((u >> 16) & 1u);
  return (unsigned short)(u >> 16);
}
static __device__ __forceinline__ __bf16 us2bf(unsigned short u) {
  union { unsigned short u; __bf16 b; } x; x.u = u; return x.b;
}

// ---------------------------------------------------------------------------
// Async global->LDS copy (16B). gfx1250 GLOBAL_LOAD_ASYNC_TO_LDS_B128 is
// ASYNCcnt-tracked and bypasses VGPRs; fall back to a regular b128 round-trip
// if this toolchain doesn't declare the builtin.
// ---------------------------------------------------------------------------
#if defined(__HIP_DEVICE_COMPILE__) && __has_builtin(__builtin_amdgcn_global_load_async_to_lds_b128)
#define ASYNC_LDS 1
#else
#define ASYNC_LDS 0
#endif

static __device__ __forceinline__ void cp16_g2l(unsigned short* lds_dst,
                                                const unsigned short* g_src) {
#if ASYNC_LDS
  __builtin_amdgcn_global_load_async_to_lds_b128(
      (g_v4i_p)(void*)g_src,
      (l_v4i_p)(void*)lds_dst,
      0, 0);
#else
  *(uint4*)lds_dst = *(const uint4*)g_src;
#endif
}
static __device__ __forceinline__ void cp16_wait() {
#if ASYNC_LDS
#if __has_builtin(__builtin_amdgcn_s_wait_asynccnt)
  __builtin_amdgcn_s_wait_asynccnt(0);
#else
  asm volatile("s_wait_asynccnt 0x0" ::: "memory");
#endif
#endif
}

// ---------------------------------------------------------------------------
// K1: transpose W_w / W_b  (256 x 40960 -> 40960 x 256) so per-feature rows
// are contiguous 1KB gathers (L2-resident, fully used cachelines).
// ---------------------------------------------------------------------------
__global__ void k_transpose(const float* __restrict__ Ww, const float* __restrict__ Wb,
                            float* __restrict__ Tw, float* __restrict__ Tb) {
  __shared__ float tile[32][33];
  const float* src = (blockIdx.z == 0) ? Ww : Wb;
  float*       dst = (blockIdx.z == 0) ? Tw : Tb;
  int f0 = blockIdx.x * 32;
  int o0 = blockIdx.y * 32;
  int tx = threadIdx.x, ty = threadIdx.y;
#pragma unroll
  for (int j = 0; j < 4; ++j) {
    int o = o0 + ty + j * 8;
    tile[ty + j * 8][tx] = src[(size_t)o * HALF_KP + (f0 + tx)];
  }
  __syncthreads();
#pragma unroll
  for (int j = 0; j < 4; ++j) {
    int f = f0 + ty + j * 8;
    dst[(size_t)f * 256 + (o0 + tx)] = tile[tx][ty + j * 8];
  }
}

// ---------------------------------------------------------------------------
// K1b: f32 -> bf16 elementwise (Wm0, Wm1)
// ---------------------------------------------------------------------------
__global__ void k_f32_to_bf16(const float* __restrict__ in,
                              unsigned short* __restrict__ out, int n) {
  int i = blockIdx.x * blockDim.x + threadIdx.x;
  if (i < n) out[i] = f2bf_u(in[i]);
}

// ---------------------------------------------------------------------------
// K2: feature transform. One 256-thread block per position. float4 scan so
// the dominant 1.34 GB read issues b128s. Index lists sorted for a
// deterministic accumulation order (ascending f, like the dense matmul).
// Multihot values are exactly 1.0, so gather-sum == the reference matmul.
// ---------------------------------------------------------------------------
static __device__ __forceinline__ void push_idx(int f, int* list, int* cnt) {
  int p = atomicAdd(cnt, 1);
  if (p < 64) list[p] = f;
}

__global__ void k_feature(const float* __restrict__ pov,
                          const float* __restrict__ white, const float* __restrict__ black,
                          const float* __restrict__ Tw, const float* __restrict__ Tb,
                          const float* __restrict__ bw, const float* __restrict__ bb,
                          float* __restrict__ base, unsigned short* __restrict__ base_h) {
  __shared__ int s_idxW[64], s_idxB[64];
  __shared__ int s_cntW, s_cntB;
  int b = blockIdx.x;
  int tid = threadIdx.x;
  if (tid == 0) { s_cntW = 0; s_cntB = 0; }
  __syncthreads();

  const float4* w4 = (const float4*)(white + (size_t)b * HALF_KP);
  const float4* b4 = (const float4*)(black + (size_t)b * HALF_KP);
  for (int t = tid; t < HALF_KP / 4; t += 256) {
    float4 v = w4[t];
    if (v.x != 0.0f) push_idx(4 * t + 0, s_idxW, &s_cntW);
    if (v.y != 0.0f) push_idx(4 * t + 1, s_idxW, &s_cntW);
    if (v.z != 0.0f) push_idx(4 * t + 2, s_idxW, &s_cntW);
    if (v.w != 0.0f) push_idx(4 * t + 3, s_idxW, &s_cntW);
    float4 u = b4[t];
    if (u.x != 0.0f) push_idx(4 * t + 0, s_idxB, &s_cntB);
    if (u.y != 0.0f) push_idx(4 * t + 1, s_idxB, &s_cntB);
    if (u.z != 0.0f) push_idx(4 * t + 2, s_idxB, &s_cntB);
    if (u.w != 0.0f) push_idx(4 * t + 3, s_idxB, &s_cntB);
  }
  __syncthreads();

  int cw = s_cntW < 64 ? s_cntW : 64;
  int cb = s_cntB < 64 ? s_cntB : 64;
  if (tid < 2) {                      // tiny insertion sorts (<=32 elems)
    int* a = (tid == 0) ? s_idxW : s_idxB;
    int  n = (tid == 0) ? cw : cb;
    for (int i = 1; i < n; ++i) {
      int v = a[i], j = i - 1;
      while (j >= 0 && a[j] > v) { a[j + 1] = a[j]; --j; }
      a[j + 1] = v;
    }
  }
  __syncthreads();

  int o = tid;                        // 0..255: one output channel per thread
  float accW = bw[o], accB = bb[o];
  for (int i = 0; i < cw; ++i) accW += Tw[(size_t)s_idxW[i] * 256 + o];
  for (int i = 0; i < cb; ++i) accB += Tb[(size_t)s_idxB[i] * 256 + o];

  bool P = (pov[b] != 0.0f);          // pov is exactly 0.0 / 1.0
  float first  = P ? accW : accB;     // base[:, 0:256]
  float second = P ? accB : accW;     // base[:, 256:512]
  first  = fmaxf(first, 0.0f);
  second = fmaxf(second, 0.0f);
  size_t r = (size_t)b * 512;
  base[r + o]         = first;
  base[r + 256 + o]   = second;
  base_h[r + o]       = f2bf_u(first);
  base_h[r + 256 + o] = f2bf_u(second);
}

// ---------------------------------------------------------------------------
// K3: value head. One wave32 per position; float4 fc0, fc1 via lane shuffles,
// fc2 via xor-tree reduction. Weights (64KB) stay L2-hot.
// ---------------------------------------------------------------------------
__global__ void k_value(const float* __restrict__ base,
                        const float* __restrict__ W0, const float* __restrict__ b0,
                        const float* __restrict__ W1, const float* __restrict__ b1,
                        const float* __restrict__ W2, const float* __restrict__ b2,
                        float* __restrict__ out) {
  int wid  = threadIdx.x >> 5;
  int lane = threadIdx.x & 31;
  int b = blockIdx.x * 8 + wid;
  const float4* x4 = (const float4*)(base + (size_t)b * 512);
  const float4* w4 = (const float4*)(W0 + (size_t)lane * 512);
  float s = b0[lane];
  for (int k = 0; k < 128; ++k) {
    float4 a = x4[k], w = w4[k];     // x4 reads broadcast across lanes
    s = fmaf(a.x, w.x, s); s = fmaf(a.y, w.y, s);
    s = fmaf(a.z, w.z, s); s = fmaf(a.w, w.w, s);
  }
  float x0 = fmaxf(s, 0.0f);

  float s1 = b1[lane];
#pragma unroll
  for (int j = 0; j < 32; ++j) s1 = fmaf(W1[lane * 32 + j], __shfl(x0, j, 32), s1);
  float x1 = fmaxf(s1, 0.0f);

  float p = W2[lane] * x1;
#pragma unroll
  for (int off = 16; off >= 1; off >>= 1) p += __shfl_xor(p, off, 32);
  if (lane == 0) out[b] = p + b2[0];
}

// ---------------------------------------------------------------------------
// WMMA bf16 A-fragment from global. ISA 7.12.2 16-bit A (16x32 MxK) layout:
//   lanes 0-15 -> row l, lanes 16-31 -> row l-16
//   lane group g: elems 0..7 = K k0+8g .. +7, elems 8..15 = +16.
// ---------------------------------------------------------------------------
static __device__ __forceinline__ v16bf load_frag_g(const unsigned short* __restrict__ M,
                                                    int ld, int m0, int k0, int lane) {
  int l = lane & 15, g = lane >> 4;
  const unsigned short* p = M + (size_t)(m0 + l) * ld + k0 + g * 8;
  v16bf a;
#pragma unroll
  for (int i = 0; i < 8; ++i) { a[i] = us2bf(p[i]); a[i + 8] = us2bf(p[16 + i]); }
  return a;
}
// Same fragment pattern out of an LDS tile row (row = n-n0, padded stride).
static __device__ __forceinline__ v16bf load_frag_l(const unsigned short* p) {
  v16bf a;
#pragma unroll
  for (int i = 0; i < 8; ++i) { a[i] = us2bf(p[i]); a[i + 8] = us2bf(p[16 + i]); }
  return a;
}

#define BPAD 8                        // +16B row pad: 528B stride, bank-skewed

// Stage a 64 x 256 bf16 B-chunk (rows n0..n0+63, k kc..kc+255) into sB.
static __device__ __forceinline__ void stage_B(unsigned short (*sB)[256 + BPAD],
                                               const unsigned short* __restrict__ W,
                                               int ldw, int n0, int kc, int tid) {
#pragma unroll
  for (int i = 0; i < 8; ++i) {
    int e = tid + i * 256;            // 2048 x 16B
    int row = e >> 5, c16 = e & 31;
    cp16_g2l(&sB[row][c16 * 8], W + (size_t)(n0 + row) * ldw + kc + c16 * 8);
  }
  cp16_wait();
  __syncthreads();
}

// ---------------------------------------------------------------------------
// K4: move head layer 0: a1 = relu(base @ Wm0.T + bm0), M=4096 N=256 K=512.
// Block = 8 waves sharing one 64-wide N-group; B staged in LDS in two
// 256-wide K chunks. Each wave owns a 16x64 tile (4 v8f accumulators).
// ---------------------------------------------------------------------------
__global__ void k_move0(const unsigned short* __restrict__ base_h, // [4096,512]
                        const unsigned short* __restrict__ Wm0h,   // [256,512]
                        const float* __restrict__ bm0,
                        unsigned short* __restrict__ a1h) {        // [4096,256]
  __shared__ unsigned short sB[64][256 + BPAD];
  int lane = threadIdx.x & 31;
  int wv   = threadIdx.x >> 5;
  int n0 = (blockIdx.x >> 5) * 64;                 // uniform per block
  int m0 = (((blockIdx.x & 31) * 8) + wv) * 16;
  int l15 = lane & 15, g = lane >> 4;
  v8f acc[4] = {};
  for (int kc = 0; kc < 512; kc += 256) {
    stage_B(sB, Wm0h, 512, n0, kc, threadIdx.x);
    for (int k0 = 0; k0 < 256; k0 += 32) {
      v16bf a = load_frag_g(base_h, 512, m0, kc + k0, lane);
#pragma unroll
      for (int t = 0; t < 4; ++t) {
        v16bf bf = load_frag_l(&sB[t * 16 + l15][k0 + g * 8]);
        acc[t] = __builtin_amdgcn_wmma_f32_16x16x32_bf16(
            false, a, false, bf, (short)0, acc[t], false, false);
      }
    }
    __syncthreads();
  }
#pragma unroll
  for (int t = 0; t < 4; ++t) {
    int n = n0 + t * 16 + l15;
    float bias = bm0[n];
#pragma unroll
    for (int r = 0; r < 8; ++r) {
      int m = m0 + r + 8 * g;                      // C/D layout: M = r + 8*g
      float v = fmaxf(acc[t][r] + bias, 0.0f);
      a1h[(size_t)m * 256 + n] = f2bf_u(v);
    }
  }
}

// ---------------------------------------------------------------------------
// K5: move head layer 1: a = relu(a1 @ Wm1.T + bm1), M=4096 N=4096 K=256.
// Whole B tile (64x256) staged once; cuts B L2 traffic 512MB -> 64MB.
// ---------------------------------------------------------------------------
__global__ void k_move1(const unsigned short* __restrict__ a1h,    // [4096,256]
                        const unsigned short* __restrict__ Wm1h,   // [4096,256]
                        const float* __restrict__ bm1,
                        float* __restrict__ out) {                 // [4096,4096]
  __shared__ unsigned short sB[64][256 + BPAD];
  int lane = threadIdx.x & 31;
  int wv   = threadIdx.x >> 5;
  int n0 = (blockIdx.x >> 5) * 64;                 // 64 N-groups, uniform/block
  int m0 = (((blockIdx.x & 31) * 8) + wv) * 16;
  int l15 = lane & 15, g = lane >> 4;
  stage_B(sB, Wm1h, 256, n0, 0, threadIdx.x);
  v8f acc[4] = {};
  for (int k0 = 0; k0 < 256; k0 += 32) {
    v16bf a = load_frag_g(a1h, 256, m0, k0, lane);
#pragma unroll
    for (int t = 0; t < 4; ++t) {
      v16bf bf = load_frag_l(&sB[t * 16 + l15][k0 + g * 8]);
      acc[t] = __builtin_amdgcn_wmma_f32_16x16x32_bf16(
          false, a, false, bf, (short)0, acc[t], false, false);
    }
  }
#pragma unroll
  for (int t = 0; t < 4; ++t) {
    int n = n0 + t * 16 + l15;
    float bias = bm1[n];
#pragma unroll
    for (int r = 0; r < 8; ++r) {
      int m = m0 + r + 8 * g;
      out[(size_t)m * MOVE_N + n] = fmaxf(acc[t][r] + bias, 0.0f);
    }
  }
}

// ---------------------------------------------------------------------------
extern "C" void kernel_launch(void* const* d_in, const int* in_sizes, int n_in,
                              void* d_out, int out_size, void* d_ws, size_t ws_size,
                              hipStream_t stream) {
  (void)in_sizes; (void)n_in; (void)out_size; (void)ws_size;
  const float* pov   = (const float*)d_in[0];
  const float* white = (const float*)d_in[1];
  const float* black = (const float*)d_in[2];
  const float* W_w   = (const float*)d_in[3];
  const float* b_w   = (const float*)d_in[4];
  const float* W_b   = (const float*)d_in[5];
  const float* b_b   = (const float*)d_in[6];
  const float* W0    = (const float*)d_in[7];
  const float* b0    = (const float*)d_in[8];
  const float* W1    = (const float*)d_in[9];
  const float* b1    = (const float*)d_in[10];
  const float* W2    = (const float*)d_in[11];
  const float* b2    = (const float*)d_in[12];
  const float* Wm0   = (const float*)d_in[13];
  const float* bm0   = (const float*)d_in[14];
  const float* Wm1   = (const float*)d_in[15];
  const float* bm1   = (const float*)d_in[16];

  float* x_out = (float*)d_out;            // [4096]
  float* a_out = (float*)d_out + BATCH;    // [4096*4096]

  // Workspace layout (~101 MB), chunks 256B-aligned by construction.
  char* w = (char*)d_ws;
  float* Tw = (float*)w;            w += (size_t)HALF_KP * 256 * 4;
  float* Tb = (float*)w;            w += (size_t)HALF_KP * 256 * 4;
  float* base = (float*)w;          w += (size_t)BATCH * 512 * 4;
  unsigned short* base_h = (unsigned short*)w; w += (size_t)BATCH * 512 * 2;
  unsigned short* Wm0h   = (unsigned short*)w; w += (size_t)256 * 512 * 2;
  unsigned short* Wm1h   = (unsigned short*)w; w += (size_t)MOVE_N * 256 * 2;
  unsigned short* a1h    = (unsigned short*)w; w += (size_t)BATCH * 256 * 2;

  k_transpose<<<dim3(HALF_KP / 32, 8, 2), dim3(32, 8, 1), 0, stream>>>(W_w, W_b, Tw, Tb);
  k_f32_to_bf16<<<(256 * 512 + 255) / 256, 256, 0, stream>>>(Wm0, Wm0h, 256 * 512);
  k_f32_to_bf16<<<(MOVE_N * 256 + 255) / 256, 256, 0, stream>>>(Wm1, Wm1h, MOVE_N * 256);
  k_feature<<<BATCH, 256, 0, stream>>>(pov, white, black, Tw, Tb, b_w, b_b, base, base_h);
  k_value<<<BATCH / 8, 256, 0, stream>>>(base, W0, b0, W1, b1, W2, b2, x_out);
  k_move0<<<128, 256, 0, stream>>>(base_h, Wm0h, bm0, a1h);
  k_move1<<<2048, 256, 0, stream>>>(a1h, Wm1h, bm1, a_out);
}
